// TBCNNCell_764504178786
// MI455X (gfx1250) — matrix operand order repaired and checked by reading, not compiled, not run on hardware
//
#include <hip/hip_runtime.h>
#include <hip/hip_bf16.h>
#include <stdint.h>

// CDNA5 / gfx1250: wave32, WMMA bf16 16x16x32, f32 accumulate.
typedef __attribute__((ext_vector_type(16))) __bf16 v16bf;
typedef __attribute__((ext_vector_type(8)))  float  v8f;

#define TM   64    // rows per workgroup
#define KTOT 768   // s_left(256) | s_right(256) | parent(256)
#define LDA  776   // A-tile row stride in bf16 (768 + 8 pad -> 4-bank rotation per row)

__device__ __forceinline__ unsigned short f32_to_bf16(float f) {
  union { float f; unsigned u; } v; v.f = f;
  unsigned r = v.u + 0x7FFFu + ((v.u >> 16) & 1u);   // round-to-nearest-even
  return (unsigned short)(r >> 16);
}
__device__ __forceinline__ unsigned pack_bf16x2(float a, float b) {
  return (unsigned)f32_to_bf16(a) | ((unsigned)f32_to_bf16(b) << 16);
}

// Build Wt[h][k] (bf16, k-stride KTOT): k<256 -> W_left[k][h], k<512 -> W_right[k-256][h],
// else W_top[k-512][h]. Transposed so each WMMA B-lane reads its 16 K values as two contiguous runs of 8.
__global__ void prep_weights_kernel(const float* __restrict__ Wl,
                                    const float* __restrict__ Wr,
                                    const float* __restrict__ Wtp,
                                    unsigned short* __restrict__ Wt) {
  int idx = blockIdx.x * blockDim.x + threadIdx.x;
  if (idx >= 256 * KTOT) return;
  int h = idx / KTOT;
  int k = idx - h * KTOT;
  float w;
  if (k < 256)      w = Wl[(size_t)k * 256 + h];
  else if (k < 512) w = Wr[(size_t)(k - 256) * 256 + h];
  else              w = Wtp[(size_t)(k - 512) * 256 + h];
  Wt[idx] = f32_to_bf16(w);
}

union Frag16 { unsigned int u[8]; v16bf v; };

__global__ __launch_bounds__(256) void tbcnn_fused_kernel(
    const float* __restrict__ child_h,    // [N,8,256] f32
    const float* __restrict__ parent_h,   // [N,256]   f32
    const unsigned short* __restrict__ Wt,// [256][768] bf16
    const float* __restrict__ b_conv,     // [256] f32
    float* __restrict__ out,              // [N,256] f32
    int N)
{
  extern __shared__ unsigned short AsLds[];   // TM x LDA bf16 = 99,328 B
  const int tid  = threadIdx.x;
  const int row0 = blockIdx.x * TM;

  // ---- Phase 1: C-reduction in fp32, pack bf16 A-tile into LDS -------------
  // unit u -> (row r = u/128, x-pair x = 2*(u%128)); consecutive tids hit
  // consecutive float2 -> fully coalesced global reads.
  for (int u = tid; u < TM * 128; u += 256) {
    int r = u >> 7;
    int x = (u & 127) << 1;
    int row = row0 + r;
    float sl0 = 0.f, sl1 = 0.f, sr0 = 0.f, sr1 = 0.f, p0 = 0.f, p1 = 0.f;
    if (row < N) {
      const float2* cp = (const float2*)(child_h + (size_t)row * 2048 + x);
#pragma unroll
      for (int c = 0; c < 8; ++c) {
        float2 h2 = cp[c * 128];
        float lw = (float)(7 - c) * (1.0f / 7.0f);
        float rw = 1.0f - lw;
        sl0 += lw * h2.x; sl1 += lw * h2.y;
        sr0 += rw * h2.x; sr1 += rw * h2.y;
      }
      float2 pp = *(const float2*)(parent_h + (size_t)row * 256 + x);
      p0 = pp.x; p1 = pp.y;
    }
    unsigned short* arow = AsLds + r * LDA;
    *(unsigned int*)(arow + x)       = pack_bf16x2(sl0, sl1);
    *(unsigned int*)(arow + 256 + x) = pack_bf16x2(sr0, sr1);
    *(unsigned int*)(arow + 512 + x) = pack_bf16x2(p0, p1);
  }
  __syncthreads();

  // ---- Phase 2: [64 x 768] x [768 x 256] bf16 WMMA GEMM --------------------
  const int wave  = tid >> 5;        // 8 waves: 2 (M) x 4 (N)
  const int lane  = tid & 31;
  const int wm    = wave >> 2;       // 0..1 -> 32-row strip
  const int wn    = wave & 3;        // 0..3 -> 64-col strip
  const int lrow  = lane & 15;
  const int khalf = lane >> 4;       // half-wave K offset (ISA fragment layout)

  v8f acc[2][4] = {};

  for (int k = 0; k < KTOT; k += 32) {
    Frag16 a[2], b[4];
    // A 16x32 bf16: lane holds K = {8*khalf + 0..7} U {16 + 8*khalf + 0..7}
#pragma unroll
    for (int mt = 0; mt < 2; ++mt) {
      const unsigned short* ap =
          AsLds + (32 * wm + 16 * mt + lrow) * LDA + k + 8 * khalf;
      *(uint4*)(&a[mt].u[0]) = *(const uint4*)(ap);        // ds_load_b128
      *(uint4*)(&a[mt].u[4]) = *(const uint4*)(ap + 16);   // ds_load_b128
    }
    // B 32x16 bf16: lane (col = lane&15) holds K = {8*khalf + 0..7} U {16 + 8*khalf + 0..7}
#pragma unroll
    for (int nt = 0; nt < 4; ++nt) {
      const unsigned short* bp =
          Wt + (size_t)(64 * wn + 16 * nt + lrow) * KTOT + k + 8 * khalf;
      *(uint4*)(&b[nt].u[0]) = *(const uint4*)(bp);        // global_load_b128
      *(uint4*)(&b[nt].u[4]) = *(const uint4*)(bp + 16);   // global_load_b128
    }
#pragma unroll
    for (int mt = 0; mt < 2; ++mt)
#pragma unroll
      for (int nt = 0; nt < 4; ++nt)
        acc[mt][nt] = __builtin_amdgcn_wmma_f32_16x16x32_bf16(
            false, a[mt].v, false, b[nt].v, (short)0, acc[mt][nt], false, false);
  }

  // ---- Epilogue: + bias, relu, store fp32 ----------------------------------
  // D layout: VGPR e holds M = e + 8*khalf, N = lane&15.
#pragma unroll
  for (int nt = 0; nt < 4; ++nt) {
    const int col = 64 * wn + 16 * nt + lrow;
    const float bias = b_conv[col];
#pragma unroll
    for (int mt = 0; mt < 2; ++mt) {
#pragma unroll
      for (int e = 0; e < 8; ++e) {
        int row = row0 + 32 * wm + 16 * mt + 8 * khalf + e;
        if (row < N) {
          float v = acc[mt][nt][e] + bias;
          out[(size_t)row * 256 + col] = v > 0.0f ? v : 0.0f;
        }
      }
    }
  }
}

extern "C" void kernel_launch(void* const* d_in, const int* in_sizes, int n_in,
                              void* d_out, int out_size, void* d_ws, size_t ws_size,
                              hipStream_t stream) {
  const float* child_h  = (const float*)d_in[0];
  const float* parent_h = (const float*)d_in[1];
  const float* W_left   = (const float*)d_in[2];
  const float* W_right  = (const float*)d_in[3];
  const float* W_top    = (const float*)d_in[4];
  const float* b_conv   = (const float*)d_in[5];
  float* out = (float*)d_out;

  const int N = in_sizes[1] / 256;               // parent_h is [N,256]
  unsigned short* Wt = (unsigned short*)d_ws;    // 256*768 bf16 = 384 KB scratch

  prep_weights_kernel<<<(256 * KTOT + 255) / 256, 256, 0, stream>>>(
      W_left, W_right, W_top, Wt);

  const int nblk = (N + TM - 1) / TM;
  const size_t lds_bytes = (size_t)TM * LDA * sizeof(unsigned short);
  tbcnn_fused_kernel<<<nblk, 256, lds_bytes, stream>>>(
      child_h, parent_h, Wt, b_conv, out, N);
}
